// TorchModel_85684597555712
// MI455X (gfx1250) — matrix operand order repaired and checked by reading, not compiled, not run on hardware
//
#include <hip/hip_runtime.h>
#include <hip/hip_bf16.h>

// ---------------------------------------------------------------------------
// BiLSTM-CRF forward for MI455X (gfx1250, wave32, WMMA 16x16x32 f16).
// V=21128, H=768, T=256, B=16, C=9.
// ---------------------------------------------------------------------------

typedef __attribute__((ext_vector_type(16))) _Float16 v16h;
typedef __attribute__((ext_vector_type(8)))  _Float16 v8h;
typedef __attribute__((ext_vector_type(8)))  float    v8f;

#define HID    768
#define GT     3072      // 4*H   (gates per direction)
#define NDIR2  6144      // 2*4*H (both directions)
#define SEQT   256
#define BATCH  16
#define NROWS  4096      // B*T
#define NCLS   9

__device__ __forceinline__ float sigm(float x) { return 1.0f / (1.0f + expf(-x)); }

// ---------------------------------------------------------------------------
__global__ void __launch_bounds__(512) k_cvt(const float* __restrict__ s,
                                             _Float16* __restrict__ d, int n)
{
    int i = blockIdx.x * blockDim.x + threadIdx.x;
    if (i < n) d[i] = (_Float16)s[i];
}

__global__ void __launch_bounds__(256) k_embed(const int* __restrict__ ids,
                                               const float* __restrict__ emb,
                                               _Float16* __restrict__ x)
{
    int row = blockIdx.x;                        // b*T + t
    int id  = ids[row];
    const float* src = emb + (size_t)id * HID;
    _Float16*    dst = x + (size_t)row * HID;
    for (int i = threadIdx.x; i < HID; i += blockDim.x)
        dst[i] = (_Float16)src[i];
}

// ---------------------------------------------------------------------------
// out[M][N] = X[M][K] @ W^T + biasA[N] + biasB[N]    (W is [N][K] row-major)
// Block = 256 thr = 8 waves, 32(M) x 256(N) per block; each wave 16x64.
// K-loop unrolled by 2 with register double-buffering so loads for k+1 are
// outstanding while WMMAs for k issue (breaks the s_wait_loadcnt 0 pattern).
__global__ void __launch_bounds__(256) k_gemm_xw(
    const _Float16* __restrict__ X, const _Float16* __restrict__ W,
    const float* __restrict__ bA, const float* __restrict__ bB,
    float* __restrict__ out, int M, int N, int K)
{
    int lane = threadIdx.x & 31;
    int wave = threadIdx.x >> 5;                 // 0..7
    int m0 = blockIdx.y * 32  + (wave >> 2) * 16;
    int n0 = blockIdx.x * 256 + (wave & 3) * 64;

    v8f z = {0, 0, 0, 0, 0, 0, 0, 0};
    v8f acc[4] = {z, z, z, z};

    // Lane-resolved base pointers.
    // A-tile (16x32 MxK): lanes 0-15 row=lane, K 0..15 ; lanes 16-31 K 16..31.
    const _Float16* xl = X + (size_t)(m0 + (lane & 15)) * K + ((lane >> 4) << 4);
    // B-tile (32x16 KxN), B(k,n)=W[n][k]: lanes 0-15 K=0..15, lanes 16-31 K=16..31.
    const _Float16* wl[4];
#pragma unroll
    for (int j = 0; j < 4; ++j)
        wl[j] = W + (size_t)(n0 + j * 16 + (lane & 15)) * K + ((lane >> 4) << 4);

    auto lda = [&](int k0) -> v16h {
        v8h lo = *(const v8h*)(xl + k0);
        v8h hi = *(const v8h*)(xl + k0 + 8);
        v16h a;
#pragma unroll
        for (int i = 0; i < 8; ++i) { a[i] = lo[i]; a[i + 8] = hi[i]; }
        return a;
    };

    v16h a0 = lda(0);
    v16h b0[4];
#pragma unroll
    for (int j = 0; j < 4; ++j) b0[j] = *(const v16h*)(wl[j]);

    for (int k0 = 0; k0 < K; k0 += 64) {
        v16h a1 = lda(k0 + 32);
        v16h b1[4];
#pragma unroll
        for (int j = 0; j < 4; ++j) b1[j] = *(const v16h*)(wl[j] + k0 + 32);
#pragma unroll
        for (int j = 0; j < 4; ++j)
            acc[j] = __builtin_amdgcn_wmma_f32_16x16x32_f16(
                false, a0, false, b0[j], (short)0, acc[j], false, false);
        if (k0 + 64 < K) {
            a0 = lda(k0 + 64);
#pragma unroll
            for (int j = 0; j < 4; ++j) b0[j] = *(const v16h*)(wl[j] + k0 + 64);
        }
#pragma unroll
        for (int j = 0; j < 4; ++j)
            acc[j] = __builtin_amdgcn_wmma_f32_16x16x32_f16(
                false, a1, false, b1[j], (short)0, acc[j], false, false);
    }

    // C/D layout: lanes 0-15 N=lane,M=r ; lanes 16-31 N=lane-16,M=r+8
    int mb = m0 + ((lane >> 4) << 3);
#pragma unroll
    for (int j = 0; j < 4; ++j) {
        int n = n0 + j * 16 + (lane & 15);
        float bs = bA[n] + bB[n];
#pragma unroll
        for (int r = 0; r < 8; ++r)
            out[(size_t)(mb + r) * N + n] = acc[j][r] + bs;
    }
}

// ---------------------------------------------------------------------------
// LSTM recurrence for one layer, both directions (grid.x = 2).
// 768 threads = 24 waves; wave w owns hidden "quads" 2w, 2w+1 (16 units each)
// across all 4 gates -> cell state c never leaves VGPRs; h lives in LDS.
// Accumulators are initialized from xw (pre-activations) so the HBM loads
// issue at step start; the *next* step's xw rows are prefetched each step,
// and the next K-group of W_hh (L2-resident) is prefetched inside the K-loop.
__global__ void __launch_bounds__(768) k_lstm(
    const float*    __restrict__ xw,    // [4096][6144]  x@Wih^T + b_ih + b_hh
    const _Float16* __restrict__ whh,   // [2][3072][768] f16
    _Float16*       __restrict__ outh)  // [4096][1536]  fwd | bwd concat
{
    int d    = blockIdx.x;
    int tid  = threadIdx.x;
    int lane = tid & 31;
    int wave = tid >> 5;                 // 0..23

    __shared__ _Float16 h_lds[BATCH * HID];      // 24 KB
    for (int i = tid; i < BATCH * HID; i += blockDim.x) h_lds[i] = (_Float16)0.0f;

    v8f z = {0, 0, 0, 0, 0, 0, 0, 0};
    v8f cst[2] = {z, z};                 // cell state, one v8f per quad

    int n  = lane & 15;
    int mb = (lane >> 4) << 3;

    // Lane-resolved W_hh tile pointers, one per (sub, gate) tile.
    const _Float16* wbase = whh + (size_t)d * GT * HID;
    const _Float16* wl[8];
#pragma unroll
    for (int sub = 0; sub < 2; ++sub) {
        int q = wave * 2 + sub;
#pragma unroll
        for (int gi = 0; gi < 4; ++gi)
            wl[sub * 4 + gi] = wbase + (size_t)(gi * HID + q * 16 + n) * HID
                             + ((lane >> 4) << 4);
    }
    // Lane-resolved A-tile pointer into LDS h.
    const _Float16* al = h_lds + (lane & 15) * HID + ((lane >> 4) << 4);

    // warm the first K-group of W_hh
#pragma unroll
    for (int i = 0; i < 8; ++i) __builtin_prefetch(wl[i], 0, 3);

    __syncthreads();
    for (int t = 0; t < SEQT; ++t) {
        int tact = d ? (SEQT - 1 - t) : t;

        // --- init accumulators from pre-activations (issues 64 loads early)
        v8f acc[8];                      // [sub*4 + gate]
#pragma unroll
        for (int sub = 0; sub < 2; ++sub) {
            int j0 = (wave * 2 + sub) * 16 + n;
#pragma unroll
            for (int r = 0; r < 8; ++r) {
                const float* rp = xw + (size_t)((mb + r) * SEQT + tact) * NDIR2
                                + (size_t)d * GT + j0;
#pragma unroll
                for (int gi = 0; gi < 4; ++gi)
                    acc[sub * 4 + gi][r] = rp[gi * HID];
            }
        }
        // --- prefetch next step's pre-activations (hides HBM latency)
        if (t + 1 < SEQT) {
            int tn = d ? (SEQT - 2 - t) : (t + 1);
#pragma unroll
            for (int r = 0; r < 8; ++r) {
                const float* rp = xw + (size_t)((mb + r) * SEQT + tn) * NDIR2
                                + (size_t)d * GT;
#pragma unroll
                for (int gi = 0; gi < 4; ++gi)
#pragma unroll
                    for (int sub = 0; sub < 2; ++sub)
                        __builtin_prefetch(rp + gi * HID + (wave * 2 + sub) * 16 + n, 0, 2);
            }
        }

        // --- h @ Whh^T : 24 K-groups, 8 WMMAs each; prefetch K+32 ahead
#pragma unroll
        for (int k0 = 0; k0 < HID; k0 += 32) {
            if (k0 + 32 < HID) {
#pragma unroll
                for (int i = 0; i < 8; ++i) __builtin_prefetch(wl[i] + k0 + 32, 0, 3);
            }
            v8h lo = *(const v8h*)(al + k0);
            v8h hi = *(const v8h*)(al + k0 + 8);
            v16h a;
#pragma unroll
            for (int i = 0; i < 8; ++i) { a[i] = lo[i]; a[i + 8] = hi[i]; }
#pragma unroll
            for (int i = 0; i < 8; ++i) {
                v16h b = *(const v16h*)(wl[i] + k0);
                acc[i] = __builtin_amdgcn_wmma_f32_16x16x32_f16(
                    false, a, false, b, (short)0, acc[i], false, false);
            }
        }

        // --- gate nonlinearities + state update (c stays in VGPRs)
        float hval[2][8];
#pragma unroll
        for (int sub = 0; sub < 2; ++sub) {
#pragma unroll
            for (int r = 0; r < 8; ++r) {
                float gI = acc[sub * 4 + 0][r];
                float gF = acc[sub * 4 + 1][r];
                float gG = acc[sub * 4 + 2][r];
                float gO = acc[sub * 4 + 3][r];
                float cn = sigm(gF) * cst[sub][r] + sigm(gI) * tanhf(gG);
                cst[sub][r] = cn;
                hval[sub][r] = sigm(gO) * tanhf(cn);
            }
        }
        __syncthreads();                 // all h_lds reads done
#pragma unroll
        for (int sub = 0; sub < 2; ++sub) {
            int j0 = (wave * 2 + sub) * 16 + n;
#pragma unroll
            for (int r = 0; r < 8; ++r) {
                int m = mb + r;
                _Float16 hv = (_Float16)hval[sub][r];
                h_lds[m * HID + j0] = hv;
                outh[(size_t)(m * SEQT + tact) * (2 * HID) + (size_t)d * HID + j0] = hv;
            }
        }
        __syncthreads();                 // new h visible to all waves
    }
}

// ---------------------------------------------------------------------------
// logits[M][9] = out1[M][1536] @ Wcls[1536][9] + bcls   (tiny: 113 MFLOP)
__global__ void __launch_bounds__(256) k_cls(
    const _Float16* __restrict__ x, const float* __restrict__ W,
    const float* __restrict__ b, float* __restrict__ out)
{
    int idx = blockIdx.x * blockDim.x + threadIdx.x;
    if (idx >= NROWS * NCLS) return;
    int row = idx / NCLS, c = idx % NCLS;
    const _Float16* xr = x + (size_t)row * (2 * HID);
    float s = b[c];
    for (int k = 0; k < 2 * HID; ++k)
        s += (float)xr[k] * W[k * NCLS + c];
    out[idx] = s;
}

// CRF NLL mean, one workgroup. alpha recursion: 144 (b,c) threads.
__global__ void __launch_bounds__(256) k_crf(
    const float* __restrict__ em,       // [B*T][9]
    const int*   __restrict__ labels,   // [B][T]
    const float* __restrict__ st, const float* __restrict__ en,
    const float* __restrict__ trans, float* __restrict__ out)
{
    __shared__ float tr[NCLS * NCLS];
    __shared__ float A0[BATCH * NCLS], A1[BATCH * NCLS];
    __shared__ float logZ[BATCH], score[BATCH];
    int tid = threadIdx.x;
    if (tid < NCLS * NCLS) tr[tid] = trans[tid];
    __syncthreads();

    if (tid < BATCH) {                   // gold-path score, one thread per batch
        int b = tid;
        const int* lab = labels + b * SEQT;
        int prev = lab[0];
        float s = st[prev] + em[(size_t)(b * SEQT) * NCLS + prev];
        int cnt = (lab[0] > -1) ? 1 : 0;
        for (int t = 1; t < SEQT; ++t) {
            int tg = lab[t];
            if (tg > -1) {
                s += tr[prev * NCLS + tg] + em[(size_t)(b * SEQT + t) * NCLS + tg];
                cnt++;
            }
            prev = tg;
        }
        s += en[lab[cnt - 1]];
        score[b] = s;
    }

    int b = tid / NCLS, c = tid % NCLS;
    bool act = tid < BATCH * NCLS;
    if (act) A0[tid] = st[c] + em[(size_t)(b * SEQT) * NCLS + c];
    __syncthreads();

    float* cur = A0; float* nxt = A1;
    for (int t = 1; t < SEQT; ++t) {
        float v = 0.0f;
        if (act) {
            float e  = em[(size_t)(b * SEQT + t) * NCLS + c];
            float mx = -3.4e38f;
            float tmp[NCLS];
#pragma unroll
            for (int k = 0; k < NCLS; ++k) {
                float x = cur[b * NCLS + k] + tr[k * NCLS + c] + e;
                tmp[k] = x; mx = fmaxf(mx, x);
            }
            float sum = 0.0f;
#pragma unroll
            for (int k = 0; k < NCLS; ++k) sum += expf(tmp[k] - mx);
            float ns = mx + logf(sum);
            v = (labels[b * SEQT + t] > -1) ? ns : cur[b * NCLS + c];
        }
        __syncthreads();
        if (act) nxt[tid] = v;
        __syncthreads();
        float* tp = cur; cur = nxt; nxt = tp;
    }

    if (act && c == 0) {
        float mx = -3.4e38f; float tmp[NCLS];
#pragma unroll
        for (int k = 0; k < NCLS; ++k) { float x = cur[b * NCLS + k] + en[k]; tmp[k] = x; mx = fmaxf(mx, x); }
        float sum = 0.0f;
#pragma unroll
        for (int k = 0; k < NCLS; ++k) sum += expf(tmp[k] - mx);
        logZ[b] = mx + logf(sum);
    }
    __syncthreads();
    if (tid == 0) {
        float s = 0.0f;
        for (int bb = 0; bb < BATCH; ++bb) s += (logZ[bb] - score[bb]);
        out[0] = s / (float)BATCH;
    }
}

// ---------------------------------------------------------------------------
extern "C" void kernel_launch(void* const* d_in, const int* in_sizes, int n_in,
                              void* d_out, int out_size, void* d_ws, size_t ws_size,
                              hipStream_t stream)
{
    (void)in_sizes; (void)n_in; (void)out_size; (void)ws_size;

    const int*   ids    = (const int*)d_in[0];
    const int*   labels = (const int*)d_in[1];
    const float* emb    = (const float*)d_in[2];
    const float* wih0   = (const float*)d_in[3];
    const float* whh0   = (const float*)d_in[4];
    const float* bih0   = (const float*)d_in[5];
    const float* bhh0   = (const float*)d_in[6];
    const float* wih1   = (const float*)d_in[7];
    const float* whh1   = (const float*)d_in[8];
    const float* bih1   = (const float*)d_in[9];
    const float* bhh1   = (const float*)d_in[10];
    const float* Wcls   = (const float*)d_in[11];
    const float* bcls   = (const float*)d_in[12];
    const float* strt   = (const float*)d_in[13];
    const float* endt   = (const float*)d_in[14];
    const float* trans  = (const float*)d_in[15];

    // workspace layout (~180 MB total)
    char* ws = (char*)d_ws;
    size_t off = 0;
    auto alloc = [&](size_t bytes) -> char* {
        char* p = ws + off;
        off += (bytes + 255) & ~(size_t)255;
        return p;
    };
    const size_t NW0  = (size_t)NDIR2 * HID;       // 6144*768
    const size_t NW1i = (size_t)NDIR2 * 2 * HID;   // 6144*1536
    _Float16* wih0h = (_Float16*)alloc(NW0 * 2);
    _Float16* whh0h = (_Float16*)alloc(NW0 * 2);
    _Float16* wih1h = (_Float16*)alloc(NW1i * 2);
    _Float16* whh1h = (_Float16*)alloc(NW0 * 2);
    _Float16* x0h   = (_Float16*)alloc((size_t)NROWS * HID * 2);
    _Float16* out0h = (_Float16*)alloc((size_t)NROWS * 2 * HID * 2);
    _Float16* out1h = (_Float16*)alloc((size_t)NROWS * 2 * HID * 2);
    float*    xwbuf = (float*)alloc((size_t)NROWS * NDIR2 * 4);   // reused by both layers
    float*    logit = (float*)alloc((size_t)NROWS * NCLS * 4);

    auto cvt = [&](const float* s, _Float16* dst, size_t n) {
        int blocks = (int)((n + 511) / 512);
        k_cvt<<<blocks, 512, 0, stream>>>(s, dst, (int)n);
    };
    cvt(wih0, wih0h, NW0);
    cvt(whh0, whh0h, NW0);
    cvt(wih1, wih1h, NW1i);
    cvt(whh1, whh1h, NW0);

    k_embed<<<NROWS, 256, 0, stream>>>(ids, emb, x0h);

    dim3 gg(NDIR2 / 256, NROWS / 32);
    // layer 0: xw = x @ Wih0^T + b_ih + b_hh ; then recurrence
    k_gemm_xw<<<gg, 256, 0, stream>>>(x0h, wih0h, bih0, bhh0, xwbuf, NROWS, NDIR2, HID);
    k_lstm<<<2, 768, 0, stream>>>(xwbuf, whh0h, out0h);
    // layer 1
    k_gemm_xw<<<gg, 256, 0, stream>>>(out0h, wih1h, bih1, bhh1, xwbuf, NROWS, NDIR2, 2 * HID);
    k_lstm<<<2, 768, 0, stream>>>(xwbuf, whh1h, out1h);

    k_cls<<<(NROWS * NCLS + 255) / 256, 256, 0, stream>>>(out1h, Wcls, bcls, logit);
    k_crf<<<1, 256, 0, stream>>>(logit, labels, strt, endt, trans, (float*)d_out);
}
